// multi_head_attention_61392262529322
// MI455X (gfx1250) — matrix operand order repaired and checked by reading, not compiled, not run on hardware
//
#include <hip/hip_runtime.h>
#include <hip/hip_bf16.h>

typedef __attribute__((ext_vector_type(16))) _Float16 v16h;
typedef __attribute__((ext_vector_type(8)))  _Float16 v8h;
typedef __attribute__((ext_vector_type(8)))  float    v8f;
typedef __attribute__((ext_vector_type(4)))  float    f32x4;

#define T_DIM   512
#define N_DIM   128
#define D_DIM   64
#define H_DIM   4
#define HID_DIM 64
#define OUT_DIM 64
#define HD      256            // H*HID
#define HSZ     (T_DIM * N_DIM * HID_DIM)   // per-head Q/K plane: 4194304 elems

union AF { v16h v; v8h h2[2]; _Float16 h[16]; };
union CF { v8f  v; float f[8]; };

static __device__ __forceinline__ v8f wmma16(v16h a, v16h b, v8f c) {
  // D = A(16x32 f16) * B(32x16 f16) + C(16x16 f32)
  return __builtin_amdgcn_wmma_f32_16x16x32_f16(false, a, false, b, (short)0, c,
                                                false, false);
}

// ---------------------------------------------------------------------------
// Kernel 0: one-time f32->f16 weight conversion.
// Wh = concat(wq,wk,wv) as [768][64] f16 ; woh = wo as [64][256] f16.
// ---------------------------------------------------------------------------
__global__ __launch_bounds__(256) void cvt_weights_kernel(
    const float* __restrict__ wq, const float* __restrict__ wk,
    const float* __restrict__ wv, const float* __restrict__ wo,
    _Float16* __restrict__ Wh, _Float16* __restrict__ woh)
{
  const int i = blockIdx.x * 256 + threadIdx.x;   // 0..65535
  if (i < 16384)      Wh[i] = (_Float16)wq[i];
  else if (i < 32768) Wh[i] = (_Float16)wk[i - 16384];
  else if (i < 49152) Wh[i] = (_Float16)wv[i - 32768];
  else                woh[i - 49152] = (_Float16)wo[i - 49152];
}

// ---------------------------------------------------------------------------
// Kernel 1: xin = x + pe ; QKV projection GEMM [65536 x 64] * [64 x 768]
// Q,K stored [h][m=t*N+n][e] (f16) -> store address linear in m.
// V stored transposed [h][n][e][t] (f16) for contiguous P*V B-fragments.
// ---------------------------------------------------------------------------
__global__ __launch_bounds__(128) void qkv_proj_kernel(
    const float* __restrict__ x,  const float* __restrict__ pe,
    const _Float16* __restrict__ Wh,
    _Float16* __restrict__ Qh, _Float16* __restrict__ Kh,
    _Float16* __restrict__ Vt)
{
  const int wave = threadIdx.x >> 5;
  const int lane = threadIdx.x & 31;
  const int lx   = lane & 15;
  const int hi   = lane >> 4;
  const int mW   = blockIdx.x * 64 + wave * 16;   // wave's 16-row tile base
  const int mRow = mW + lx;                       // A-fragment row this lane owns

  // A fragments (xin rows), K-halves [0,32) and [32,64)
  AF a0, a1;
  {
    const int base = mRow * D_DIM;
#pragma unroll
    for (int kk = 0; kk < 2; ++kk) {
      AF& a = kk ? a1 : a0;
      const int k0 = kk * 32;
      f32x4 x0 = ((const f32x4*)(x  + base + k0      + hi * 8))[0];
      f32x4 x1 = ((const f32x4*)(x  + base + k0      + hi * 8))[1];
      f32x4 x2 = ((const f32x4*)(x  + base + k0 + 16 + hi * 8))[0];
      f32x4 x3 = ((const f32x4*)(x  + base + k0 + 16 + hi * 8))[1];
      f32x4 p0 = ((const f32x4*)(pe + base + k0      + hi * 8))[0];
      f32x4 p1 = ((const f32x4*)(pe + base + k0      + hi * 8))[1];
      f32x4 p2 = ((const f32x4*)(pe + base + k0 + 16 + hi * 8))[0];
      f32x4 p3 = ((const f32x4*)(pe + base + k0 + 16 + hi * 8))[1];
#pragma unroll
      for (int i = 0; i < 4; ++i) {
        a.h[i]      = (_Float16)(x0[i] + p0[i]);
        a.h[4 + i]  = (_Float16)(x1[i] + p1[i]);
        a.h[8 + i]  = (_Float16)(x2[i] + p2[i]);
        a.h[12 + i] = (_Float16)(x3[i] + p3[i]);
      }
    }
  }

  // shared tile computation: D(16x16) = Xin_tile * Wh[jc..jc+16)^T
  auto compute_tile = [&](int jc) {
    const _Float16* whrow = Wh + (size_t)(jc + lx) * D_DIM;
    AF b0, b1;                                // B[k][j] = Wh[j][k] (contig in k)
    b0.h2[0] = *(const v8h*)(whrow      + hi * 8);
    b0.h2[1] = *(const v8h*)(whrow + 16 + hi * 8);
    b1.h2[0] = *(const v8h*)(whrow + 32 + hi * 8);
    b1.h2[1] = *(const v8h*)(whrow + 48 + hi * 8);
    CF c;
#pragma unroll
    for (int r = 0; r < 8; ++r) c.f[r] = 0.f;
    c.v = wmma16(a0.v, b0.v, c.v);
    c.v = wmma16(a1.v, b1.v, c.v);
    return c;
  };

  // ---- Q tiles: jcol in [0,256) ----
  for (int jt = 0; jt < 16; ++jt) {
    const int jc = jt * 16;
    CF c = compute_tile(jc);
    const int jcol = jc + lx;
    _Float16* base = Qh + (size_t)(jcol >> 6) * HSZ + (jcol & 63);
#pragma unroll
    for (int r = 0; r < 8; ++r)
      base[(size_t)(mW + r + 8 * hi) * HID_DIM] = (_Float16)c.f[r];
  }
  // ---- K tiles: jcol in [256,512) ----
  for (int jt = 16; jt < 32; ++jt) {
    const int jc = jt * 16;
    CF c = compute_tile(jc);
    const int jcol = (jc & 255) + lx;
    _Float16* base = Kh + (size_t)(jcol >> 6) * HSZ + (jcol & 63);
#pragma unroll
    for (int r = 0; r < 8; ++r)
      base[(size_t)(mW + r + 8 * hi) * HID_DIM] = (_Float16)c.f[r];
  }
  // ---- V tiles: jcol in [512,768), transposed store ----
  for (int jt = 32; jt < 48; ++jt) {
    const int jc = jt * 16;
    CF c = compute_tile(jc);
    const int jcol = jc + lx;
    const int h = (jcol >> 6) & 3, e = jcol & 63;
    _Float16* vb = Vt + ((size_t)h * N_DIM * HID_DIM + e) * T_DIM;
#pragma unroll
    for (int r = 0; r < 8; ++r) {
      const int m = mW + r + 8 * hi;
      const int t = m >> 7, n = m & 127;
      vb[(size_t)n * (HID_DIM * T_DIM) + t] = (_Float16)c.f[r];
    }
  }
}

// ---------------------------------------------------------------------------
// Kernel 2: causal flash attention per (h, n). One wave = 16 query rows.
// grid = (T/128, N, H), block = 256 (8 waves).
// ---------------------------------------------------------------------------
__global__ __launch_bounds__(256) void attn_kernel(
    const _Float16* __restrict__ Qh, const _Float16* __restrict__ Kh,
    const _Float16* __restrict__ Vt, _Float16* __restrict__ heads)
{
  __shared__ alignas(16) _Float16 pbuf[8][16][32];   // per-wave P bounce tile

  const int wave = threadIdx.x >> 5;
  const int lane = threadIdx.x & 31;
  const int lx   = lane & 15;
  const int hi   = lane >> 4;
  const int n    = blockIdx.y;
  const int h    = blockIdx.z;
  const int qWaveBase = blockIdx.x * 128 + wave * 16;
  const int RSTRIDE = N_DIM * HID_DIM;     // row stride in [h][t*N+n][e] layout

  const _Float16* Qbase = Qh + (size_t)h * HSZ + (size_t)n * HID_DIM;
  const _Float16* Kbase = Kh + (size_t)h * HSZ + (size_t)n * HID_DIM;
  const _Float16* Vbase = Vt + (size_t)(h * N_DIM + n) * HID_DIM * T_DIM;

  // Q fragments, pre-scaled by 1/sqrt(HID) = 0.125
  AF qa0, qa1;
  {
    const _Float16* qrow = Qbase + (size_t)(qWaveBase + lx) * RSTRIDE;
    qa0.h2[0] = *(const v8h*)(qrow      + hi * 8);
    qa0.h2[1] = *(const v8h*)(qrow + 16 + hi * 8);
    qa1.h2[0] = *(const v8h*)(qrow + 32 + hi * 8);
    qa1.h2[1] = *(const v8h*)(qrow + 48 + hi * 8);
    const _Float16 s = (_Float16)0.125f;
#pragma unroll
    for (int i = 0; i < 16; ++i) { qa0.h[i] *= s; qa1.h[i] *= s; }
  }

  CF acc[4];
#pragma unroll
  for (int ot = 0; ot < 4; ++ot)
#pragma unroll
    for (int r = 0; r < 8; ++r) acc[ot].f[r] = 0.f;
  float rowm[8], rowl[8];
#pragma unroll
  for (int r = 0; r < 8; ++r) { rowm[r] = -1e30f; rowl[r] = 0.f; }

  const int nkt = (qWaveBase + 47) >> 5;     // 32-wide key tiles covering s <= q_max
  for (int kt = 0; kt < nkt; ++kt) {
    const int sBase = kt * 32;

    if (kt + 1 < nkt) {                      // global_prefetch_b8 next key tile
      __builtin_prefetch(Kbase + (size_t)(sBase + 32 + lx) * RSTRIDE, 0, 0);
      __builtin_prefetch(Kbase + (size_t)(sBase + 48 + lx) * RSTRIDE, 0, 0);
    }

    // S = Q * K^T for two 16-wide s-subtiles
    CF c[2];
#pragma unroll
    for (int ss = 0; ss < 2; ++ss) {
      const _Float16* krow = Kbase + (size_t)(sBase + ss * 16 + lx) * RSTRIDE;
      AF kb0, kb1;                           // B[e][s] = K[s][e], contig in e
      kb0.h2[0] = *(const v8h*)(krow      + hi * 8);
      kb0.h2[1] = *(const v8h*)(krow + 16 + hi * 8);
      kb1.h2[0] = *(const v8h*)(krow + 32 + hi * 8);
      kb1.h2[1] = *(const v8h*)(krow + 48 + hi * 8);
#pragma unroll
      for (int r = 0; r < 8; ++r) c[ss].f[r] = 0.f;
      c[ss].v = wmma16(qa0.v, kb0.v, c[ss].v);
      c[ss].v = wmma16(qa1.v, kb1.v, c[ss].v);
    }

    // causal mask + online softmax (row = 16 lanes sharing lane>>4 group)
#pragma unroll
    for (int r = 0; r < 8; ++r) {
      const int q  = qWaveBase + r + 8 * hi;
      const int s0 = sBase + lx;
      if (s0 > q)      c[0].f[r] = -1e30f;
      if (s0 + 16 > q) c[1].f[r] = -1e30f;
      float tm = fmaxf(c[0].f[r], c[1].f[r]);
#pragma unroll
      for (int off = 8; off >= 1; off >>= 1)
        tm = fmaxf(tm, __shfl_xor(tm, off, 32));
      const float mnew = fmaxf(rowm[r], tm);
      const float corr = __expf(rowm[r] - mnew);
      const float p0 = __expf(c[0].f[r] - mnew);
      const float p1 = __expf(c[1].f[r] - mnew);
      float ps = p0 + p1;
#pragma unroll
      for (int off = 8; off >= 1; off >>= 1)
        ps += __shfl_xor(ps, off, 32);
      rowl[r] = rowl[r] * corr + ps;
      rowm[r] = mnew;
#pragma unroll
      for (int ot = 0; ot < 4; ++ot) acc[ot].f[r] *= corr;
      // C-layout -> LDS so we can reread P in A-fragment layout
      pbuf[wave][r + 8 * hi][lx]      = (_Float16)p0;
      pbuf[wave][r + 8 * hi][lx + 16] = (_Float16)p1;
    }
    asm volatile("s_wait_dscnt 0" ::: "memory");   // same-wave DS is in-order

    AF pa;                                         // P as 16x32 A fragment
    pa.h2[0] = *(const v8h*)(&pbuf[wave][lx][hi * 8]);
    pa.h2[1] = *(const v8h*)(&pbuf[wave][lx][16 + hi * 8]);

    // O += P * V  (V transposed: B[s][o] read contiguously along s)
#pragma unroll
    for (int ot = 0; ot < 4; ++ot) {
      const _Float16* vrow = Vbase + (size_t)(ot * 16 + lx) * T_DIM + sBase;
      AF vb;
      vb.h2[0] = *(const v8h*)(vrow + hi * 8);
      vb.h2[1] = *(const v8h*)(vrow + 16 + hi * 8);
      acc[ot].v = wmma16(pa.v, vb.v, acc[ot].v);
    }
  }

  // normalize and store heads [t][n][h*64 + o] (f16)
#pragma unroll
  for (int r = 0; r < 8; ++r) {
    const int q   = qWaveBase + r + 8 * hi;
    const float inv = 1.f / rowl[r];
    _Float16* orow = heads + (size_t)(q * N_DIM + n) * HD + h * HID_DIM;
#pragma unroll
    for (int ot = 0; ot < 4; ++ot)
      orow[ot * 16 + lx] = (_Float16)(acc[ot].f[r] * inv);
  }
}

// ---------------------------------------------------------------------------
// Kernel 3: output projection [65536 x 256] * [256 x 64] -> f32 d_out
// ---------------------------------------------------------------------------
__global__ __launch_bounds__(256) void out_proj_kernel(
    const _Float16* __restrict__ heads, const _Float16* __restrict__ woh,
    float* __restrict__ out)
{
  const int wave  = threadIdx.x >> 5;
  const int lane  = threadIdx.x & 31;
  const int lx    = lane & 15;
  const int hi    = lane >> 4;
  const int mBase = blockIdx.x * 128 + wave * 16;

  CF c[4];
#pragma unroll
  for (int jt = 0; jt < 4; ++jt)
#pragma unroll
    for (int r = 0; r < 8; ++r) c[jt].f[r] = 0.f;

  const _Float16* arow = heads + (size_t)(mBase + lx) * HD;

#pragma unroll
  for (int kb = 0; kb < 8; ++kb) {           // K = 256 in 8 steps of 32
    AF a;
    a.h2[0] = *(const v8h*)(arow + kb * 32      + hi * 8);
    a.h2[1] = *(const v8h*)(arow + kb * 32 + 16 + hi * 8);
#pragma unroll
    for (int jt = 0; jt < 4; ++jt) {
      const _Float16* wrow = woh + (size_t)(jt * 16 + lx) * HD + kb * 32 + hi * 8;
      AF b;                                  // B[k][j] = wo[j][k], contig in k
      b.h2[0] = *(const v8h*)(wrow);
      b.h2[1] = *(const v8h*)(wrow + 16);
      c[jt].v = wmma16(a.v, b.v, c[jt].v);
    }
  }

#pragma unroll
  for (int jt = 0; jt < 4; ++jt)
#pragma unroll
    for (int r = 0; r < 8; ++r) {
      const int m = mBase + r + 8 * hi;
      out[(size_t)m * OUT_DIM + jt * 16 + lx] = c[jt].f[r];
    }
}

// ---------------------------------------------------------------------------
extern "C" void kernel_launch(void* const* d_in, const int* in_sizes, int n_in,
                              void* d_out, int out_size, void* d_ws, size_t ws_size,
                              hipStream_t stream) {
  (void)in_sizes; (void)n_in; (void)out_size; (void)ws_size;
  const float* x  = (const float*)d_in[0];
  const float* pe = (const float*)d_in[1];
  const float* wq = (const float*)d_in[2];
  const float* wk = (const float*)d_in[3];
  const float* wv = (const float*)d_in[4];
  const float* wo = (const float*)d_in[5];
  float* out = (float*)d_out;

  const size_t HQ = (size_t)H_DIM * HSZ;        // 16M elems (32MB f16)
  _Float16* Qh    = (_Float16*)d_ws;
  _Float16* Kh    = Qh + HQ;
  _Float16* Vt    = Kh + HQ;
  _Float16* heads = Vt + HQ;
  _Float16* Wh    = heads + HQ;                 // 768*64  f16
  _Float16* woh   = Wh + 768 * 64;              // 64*256  f16

  cvt_weights_kernel<<<dim3(256), dim3(256), 0, stream>>>(wq, wk, wv, wo, Wh, woh);
  qkv_proj_kernel<<<dim3(1024), dim3(128), 0, stream>>>(x, pe, Wh, Qh, Kh, Vt);
  attn_kernel<<<dim3(T_DIM / 128, N_DIM, H_DIM), dim3(256), 0, stream>>>(
      Qh, Kh, Vt, heads);
  out_proj_kernel<<<dim3(512), dim3(256), 0, stream>>>(heads, woh, out);
}
